// GraphGCNNet_11081015623737
// MI455X (gfx1250) — compile-verified, matrix operand order checked by reading, not run on hardware
//
#include <hip/hip_runtime.h>
#include <hip/hip_bf16.h>
#include <cstdint>
#include <cstddef>

// ---------------------------------------------------------------------------
// GCN forward for MI455X (gfx1250, wave32, WMMA bf16 path)
// ---------------------------------------------------------------------------

typedef __bf16 bf16;
typedef __attribute__((ext_vector_type(16))) __bf16 v16bf;
typedef __attribute__((ext_vector_type(8)))  __bf16 v8bf;
typedef __attribute__((ext_vector_type(8)))  float  v8f;

#define NNODES 50000
#define NEDGES 1600000
#define NFEATC 512
#define H2C    512
#define H3C    768
#define NCLS   64
#define MM_PARTS 1024

// ---------------------------------------------------------------------------
// fp32 -> bf16 elementwise convert
// ---------------------------------------------------------------------------
__global__ void cvt_bf16_kernel(const float* __restrict__ in, bf16* __restrict__ out, int n) {
    int i = blockIdx.x * blockDim.x + threadIdx.x;
    if (i < n) out[i] = (bf16)in[i];
}

// fp32 W[K,N] -> bf16 Wt[N,K] (transpose + convert; weights are tiny)
__global__ void cvt_transpose_kernel(const float* __restrict__ W, bf16* __restrict__ Wt,
                                     int K, int N) {
    int i = blockIdx.x * blockDim.x + threadIdx.x;
    if (i < K * N) {
        int k = i / N, n = i % N;
        Wt[(size_t)n * K + k] = (bf16)W[i];
    }
}

// ---------------------------------------------------------------------------
// WMMA bf16 GEMM:  C[M,N] = A[M,K](bf16,row) @ Bt[N,K](bf16,row-of-transpose)
// Block: 256 threads = 8 waves; block tile 256(M) x 64(N).
// Per wave: 2 M-tiles x 4 N-tiles = 8 accumulators.
// Ping-pong register double buffering with *indexed* addressing (base + k0,
// bases never mutated) so clang keeps addrspace(1) -> global_load_b128 (not
// flat_load; flat would also burn DScnt). Loads for one buffer are in flight
// while the other buffer's 8 WMMAs issue. Requires K % 64 == 0 (K = 512/768).
// A-fragment per-lane layout (16-bit 16x32): lanes 0-15 row=r K={0..7,16..23},
// lanes 16-31 row=r K={8..15,24..31} -> two global_load_b128 per fragment.
// ---------------------------------------------------------------------------
__device__ __forceinline__ v16bf load_frag32(const bf16* p) {
    v8bf lo = *(const v8bf*)(p);        // K = k0 + half*8 .. +7
    v8bf hi = *(const v8bf*)(p + 16);   // K = k0 + 16 + half*8 .. +7
    return __builtin_shufflevector(lo, hi, 0,1,2,3,4,5,6,7,8,9,10,11,12,13,14,15);
}

template <bool HAS_BIAS, bool HAS_ADD, bool WF32, bool WBF16>
__global__ void __launch_bounds__(256)
gemm_wmma_bf16(const bf16* __restrict__ A, const bf16* __restrict__ Bt,
               const float* __restrict__ bias,   // [N]   (HAS_BIAS)
               const float* __restrict__ add,    // [M,N] (HAS_ADD): += 0.5*(add+bias2)
               const float* __restrict__ bias2,  // [N]   (HAS_ADD)
               float* __restrict__ Cf, bf16* __restrict__ Cb,
               int M, int N, int K) {
    const int lane = threadIdx.x & 31;
    const int wave = threadIdx.x >> 5;
    const int half = lane >> 4;       // 0: lanes 0-15, 1: lanes 16-31
    const int r    = lane & 15;

    const int m_base = blockIdx.y * 256 + wave * 32;
    const int n_base = blockIdx.x * 64;

    const bf16* aptr[2];
#pragma unroll
    for (int mt = 0; mt < 2; ++mt) {
        int arow = m_base + mt * 16 + r;
        if (arow >= M) arow = M - 1;  // clamp (store is masked below)
        aptr[mt] = A + (size_t)arow * K + half * 8;
    }

    const bf16* bptr[4];
#pragma unroll
    for (int nt = 0; nt < 4; ++nt) {
        int brow = n_base + nt * 16 + r;   // N is a multiple of 64 -> in range
        bptr[nt] = Bt + (size_t)brow * K + half * 8;
    }

    v8f acc[2][4];
#pragma unroll
    for (int mt = 0; mt < 2; ++mt)
#pragma unroll
        for (int nt = 0; nt < 4; ++nt) acc[mt][nt] = v8f(0.f);

    v16bf a0[2], b0[4], a1[2], b1[4];

    // Prologue: K-step 0 into buffer 0
#pragma unroll
    for (int mt = 0; mt < 2; ++mt) a0[mt] = load_frag32(aptr[mt]);
#pragma unroll
    for (int nt = 0; nt < 4; ++nt) b0[nt] = load_frag32(bptr[nt]);

    int k0 = 32;
    for (; k0 + 32 < K; k0 += 64) {
        // load step at k0 into buffer 1
#pragma unroll
        for (int mt = 0; mt < 2; ++mt) a1[mt] = load_frag32(aptr[mt] + k0);
#pragma unroll
        for (int nt = 0; nt < 4; ++nt) b1[nt] = load_frag32(bptr[nt] + k0);

        // compute on buffer 0
#pragma unroll
        for (int mt = 0; mt < 2; ++mt)
#pragma unroll
            for (int nt = 0; nt < 4; ++nt)
                acc[mt][nt] = __builtin_amdgcn_wmma_f32_16x16x32_bf16(
                    false, a0[mt], false, b0[nt], (short)0, acc[mt][nt], false, false);

        // load step at k0+32 into buffer 0
#pragma unroll
        for (int mt = 0; mt < 2; ++mt) a0[mt] = load_frag32(aptr[mt] + k0 + 32);
#pragma unroll
        for (int nt = 0; nt < 4; ++nt) b0[nt] = load_frag32(bptr[nt] + k0 + 32);

        // compute on buffer 1
#pragma unroll
        for (int mt = 0; mt < 2; ++mt)
#pragma unroll
            for (int nt = 0; nt < 4; ++nt)
                acc[mt][nt] = __builtin_amdgcn_wmma_f32_16x16x32_bf16(
                    false, a1[mt], false, b1[nt], (short)0, acc[mt][nt], false, false);
    }

    // Epilogue: buffer 0 holds step K-64; final step (K-32) into buffer 1
#pragma unroll
    for (int mt = 0; mt < 2; ++mt) a1[mt] = load_frag32(aptr[mt] + k0);
#pragma unroll
    for (int nt = 0; nt < 4; ++nt) b1[nt] = load_frag32(bptr[nt] + k0);

#pragma unroll
    for (int mt = 0; mt < 2; ++mt)
#pragma unroll
        for (int nt = 0; nt < 4; ++nt)
            acc[mt][nt] = __builtin_amdgcn_wmma_f32_16x16x32_bf16(
                false, a0[mt], false, b0[nt], (short)0, acc[mt][nt], false, false);
#pragma unroll
    for (int mt = 0; mt < 2; ++mt)
#pragma unroll
        for (int nt = 0; nt < 4; ++nt)
            acc[mt][nt] = __builtin_amdgcn_wmma_f32_16x16x32_bf16(
                false, a1[mt], false, b1[nt], (short)0, acc[mt][nt], false, false);

    // C/D layout: VGPR v, lanes 0-15 -> M=v, lanes 16-31 -> M=v+8; N = lane&15
#pragma unroll
    for (int mt = 0; mt < 2; ++mt) {
#pragma unroll
        for (int nt = 0; nt < 4; ++nt) {
            const int col = n_base + nt * 16 + r;
#pragma unroll
            for (int v = 0; v < 8; ++v) {
                int rowm = m_base + mt * 16 + half * 8 + v;
                if (rowm < M) {
                    size_t idx = (size_t)rowm * N + col;
                    float val = acc[mt][nt][v];
                    if (HAS_BIAS) val += bias[col];
                    if (HAS_ADD)  val += 0.5f * (add[idx] + bias2[col]);
                    if (WF32)  Cf[idx] = val;
                    if (WBF16) Cb[idx] = (bf16)val;
                }
            }
        }
    }
}

// ---------------------------------------------------------------------------
// SpMM: G[row[e]] += val[e] * T[col[e]]   (block per edge, bf16 gather keeps
// the 77MB source L2-resident; fp32 global_atomic_add_f32 accumulate)
// ---------------------------------------------------------------------------
__global__ void spmm_bf16_kernel(const int* __restrict__ rows, const int* __restrict__ cols,
                                 const float* __restrict__ vals, const bf16* __restrict__ T,
                                 float* __restrict__ G, int E, int F) {
    for (int e = blockIdx.x; e < E; e += gridDim.x) {
        const int   r = rows[e];
        const int   c = cols[e];
        const float v = vals[e];
        const v8bf* src = (const v8bf*)(T + (size_t)c * F);
        float* dst = G + (size_t)r * F;
        for (int f8 = threadIdx.x; f8 < (F >> 3); f8 += blockDim.x) {
            v8bf t = src[f8];
            int f = f8 << 3;
#pragma unroll
            for (int j = 0; j < 8; ++j)
                atomicAdd(dst + f + j, v * (float)t[j]);
        }
    }
}

// g = relu(acc + bias) -> bf16
__global__ void bias_relu_cvt_kernel(const float* __restrict__ acc,
                                     const float* __restrict__ bias,
                                     bf16* __restrict__ out, size_t total, int cols) {
    size_t i = (size_t)blockIdx.x * blockDim.x + threadIdx.x;
    if (i < total) {
        int c = (int)(i % cols);
        float v = acc[i] + bias[c];
        out[i] = (bf16)fmaxf(v, 0.0f);
    }
}

// ---------------------------------------------------------------------------
// Deterministic two-pass global min/max + normalize
// ---------------------------------------------------------------------------
__global__ void minmax_partial_kernel(const float* __restrict__ x, int n,
                                      float* __restrict__ pmin, float* __restrict__ pmax) {
    __shared__ float smin[256], smax[256];
    float mn = 3.402823466e38f, mx = -3.402823466e38f;
    for (int i = blockIdx.x * blockDim.x + threadIdx.x; i < n; i += gridDim.x * blockDim.x) {
        float v = x[i];
        mn = fminf(mn, v);
        mx = fmaxf(mx, v);
    }
    int t = threadIdx.x;
    smin[t] = mn; smax[t] = mx;
    __syncthreads();
    for (int s = 128; s > 0; s >>= 1) {
        if (t < s) {
            smin[t] = fminf(smin[t], smin[t + s]);
            smax[t] = fmaxf(smax[t], smax[t + s]);
        }
        __syncthreads();
    }
    if (t == 0) { pmin[blockIdx.x] = smin[0]; pmax[blockIdx.x] = smax[0]; }
}

__global__ void minmax_final_kernel(const float* __restrict__ pmin,
                                    const float* __restrict__ pmax,
                                    int parts, float* __restrict__ mm) {
    __shared__ float smin[256], smax[256];
    int t = threadIdx.x;
    float mn = 3.402823466e38f, mx = -3.402823466e38f;
    for (int i = t; i < parts; i += 256) {
        mn = fminf(mn, pmin[i]);
        mx = fmaxf(mx, pmax[i]);
    }
    smin[t] = mn; smax[t] = mx;
    __syncthreads();
    for (int s = 128; s > 0; s >>= 1) {
        if (t < s) {
            smin[t] = fminf(smin[t], smin[t + s]);
            smax[t] = fmaxf(smax[t], smax[t + s]);
        }
        __syncthreads();
    }
    if (t == 0) { mm[0] = smin[0]; mm[1] = smax[0]; }
}

__global__ void normalize_kernel(const float* __restrict__ in, const float* __restrict__ mm,
                                 float* __restrict__ out, int n) {
    int i = blockIdx.x * blockDim.x + threadIdx.x;
    if (i < n) {
        float mn = mm[0], mx = mm[1];
        float inv = 2.0f / (mx - mn);
        out[i] = (in[i] - mn) * inv - 1.0f;
    }
}

// ---------------------------------------------------------------------------
// Host launcher
// ---------------------------------------------------------------------------
static inline size_t align256(size_t x) { return (x + 255) & ~(size_t)255; }

extern "C" void kernel_launch(void* const* d_in, const int* in_sizes, int n_in,
                              void* d_out, int out_size, void* d_ws, size_t ws_size,
                              hipStream_t stream) {
    (void)in_sizes; (void)n_in; (void)out_size; (void)ws_size;

    const float* x     = (const float*)d_in[0];
    const int*   arow  = (const int*)  d_in[1];
    const int*   acol  = (const int*)  d_in[2];
    const float* aval  = (const float*)d_in[3];
    const float* W_org = (const float*)d_in[4];
    const float* b_org = (const float*)d_in[5];
    const float* W_gc1 = (const float*)d_in[6];
    const float* b_gc1 = (const float*)d_in[7];
    const float* W_gc2 = (const float*)d_in[8];
    const float* b_gc2 = (const float*)d_in[9];
    const float* W_l3  = (const float*)d_in[10];
    const float* b_l3  = (const float*)d_in[11];
    const float* W_out = (const float*)d_in[12];
    const float* b_out = (const float*)d_in[13];
    float* out = (float*)d_out;

    // Workspace carve-out
    char* base = (char*)d_ws;
    size_t off = 0;
    auto carve = [&](size_t bytes) -> void* {
        void* r = base + off;
        off += align256(bytes);
        return r;
    };
    bf16*  x_bf   = (bf16*) carve((size_t)NNODES * NFEATC * 2);
    bf16*  h_bf   = (bf16*) carve((size_t)NNODES * H2C * 2);
    bf16*  g_bf   = (bf16*) carve((size_t)NNODES * H3C * 2);
    bf16*  z_bf   = (bf16*) carve((size_t)NNODES * H3C * 2);
    bf16*  t_bf   = (bf16*) carve((size_t)NNODES * H3C * 2);   // SpMM gather src (L2-resident)
    float* gacc   = (float*)carve((size_t)NNODES * H3C * 4);   // SpMM fp32 accumulator
    float* outraw = (float*)carve((size_t)NNODES * NCLS * 4);
    bf16*  WorgT  = (bf16*) carve((size_t)NFEATC * H2C * 2);
    bf16*  Wgc1T  = (bf16*) carve((size_t)H2C * H3C * 2);
    bf16*  Wgc2T  = (bf16*) carve((size_t)H3C * H3C * 2);
    bf16*  Wl3T   = (bf16*) carve((size_t)H2C * H3C * 2);
    bf16*  WoutT  = (bf16*) carve((size_t)H3C * NCLS * 2);
    float* pmin   = (float*)carve(MM_PARTS * 4);
    float* pmax   = (float*)carve(MM_PARTS * 4);
    float* mm     = (float*)carve(2 * 4);

    const int mBlocks = (NNODES + 255) / 256;

    // 1) Convert inputs / weights to bf16 (weights transposed to [N,K])
    {
        int n = NNODES * NFEATC;
        cvt_bf16_kernel<<<(n + 255) / 256, 256, 0, stream>>>(x, x_bf, n);
    }
    cvt_transpose_kernel<<<(NFEATC * H2C + 255) / 256, 256, 0, stream>>>(W_org, WorgT, NFEATC, H2C);
    cvt_transpose_kernel<<<(H2C * H3C + 255) / 256, 256, 0, stream>>>(W_gc1, Wgc1T, H2C, H3C);
    cvt_transpose_kernel<<<(H3C * H3C + 255) / 256, 256, 0, stream>>>(W_gc2, Wgc2T, H3C, H3C);
    cvt_transpose_kernel<<<(H2C * H3C + 255) / 256, 256, 0, stream>>>(W_l3, Wl3T, H2C, H3C);
    cvt_transpose_kernel<<<(H3C * NCLS + 255) / 256, 256, 0, stream>>>(W_out, WoutT, H3C, NCLS);

    // 2) h = x @ W_org + b_org  -> bf16 (only ever consumed by GEMMs)
    gemm_wmma_bf16<true, false, false, true>
        <<<dim3(H2C / 64, mBlocks), 256, 0, stream>>>(
            x_bf, WorgT, b_org, nullptr, nullptr, nullptr, h_bf, NNODES, H2C, NFEATC);

    // 3) t = h @ W_gc1 -> bf16 (SpMM gather operand)
    gemm_wmma_bf16<false, false, false, true>
        <<<dim3(H3C / 64, mBlocks), 256, 0, stream>>>(
            h_bf, Wgc1T, nullptr, nullptr, nullptr, nullptr, t_bf, NNODES, H3C, H2C);

    // 4) g = relu(spmm(t) + b_gc1) -> bf16
    hipMemsetAsync(gacc, 0, (size_t)NNODES * H3C * 4, stream);
    spmm_bf16_kernel<<<NEDGES, 96, 0, stream>>>(arow, acol, aval, t_bf, gacc, NEDGES, H3C);
    {
        size_t total = (size_t)NNODES * H3C;
        bias_relu_cvt_kernel<<<(unsigned)((total + 255) / 256), 256, 0, stream>>>(
            gacc, b_gc1, g_bf, total, H3C);
    }

    // 5) t = g @ W_gc2 -> bf16; g2 = spmm(t)
    gemm_wmma_bf16<false, false, false, true>
        <<<dim3(H3C / 64, mBlocks), 256, 0, stream>>>(
            g_bf, Wgc2T, nullptr, nullptr, nullptr, nullptr, t_bf, NNODES, H3C, H3C);
    hipMemsetAsync(gacc, 0, (size_t)NNODES * H3C * 4, stream);
    spmm_bf16_kernel<<<NEDGES, 96, 0, stream>>>(arow, acol, aval, t_bf, gacc, NEDGES, H3C);

    // 6) z = (h @ W_l3 + b_l3) + 0.5*(g2 + b_gc2) -> bf16
    gemm_wmma_bf16<true, true, false, true>
        <<<dim3(H3C / 64, mBlocks), 256, 0, stream>>>(
            h_bf, Wl3T, b_l3, gacc, b_gc2, nullptr, z_bf, NNODES, H3C, H2C);

    // 7) out_raw = z @ W_out + b_out (fp32)
    gemm_wmma_bf16<true, false, true, false>
        <<<dim3(NCLS / 64, mBlocks), 256, 0, stream>>>(
            z_bf, WoutT, b_out, nullptr, nullptr, outraw, nullptr, NNODES, NCLS, H3C);

    // 8) global min/max normalize to [-1, 1]
    const int nOut = NNODES * NCLS;
    minmax_partial_kernel<<<MM_PARTS, 256, 0, stream>>>(outraw, nOut, pmin, pmax);
    minmax_final_kernel<<<1, 256, 0, stream>>>(pmin, pmax, MM_PARTS, mm);
    normalize_kernel<<<(nOut + 255) / 256, 256, 0, stream>>>(outraw, mm, out, nOut);
}
